// GNNCrossAttentionModel_12128987643945
// MI455X (gfx1250) — matrix operand order repaired and checked by reading, not compiled, not run on hardware
//
#include <hip/hip_runtime.h>
#include <hip/hip_fp16.h>

// ---------------------------------------------------------------------------
// GNN + cross-attention forward for gfx1250 (MI455X), wave32 + WMMA f16.
// ---------------------------------------------------------------------------

typedef __attribute__((ext_vector_type(16))) _Float16 v16h;
typedef __attribute__((ext_vector_type(8)))  float    v8f;

static __device__ __forceinline__ v8f wmma32(v16h a, v16h b, v8f c) {
  // D = A(16x32 f16) * B(32x16 f16) + C(16x16 f32)
  return __builtin_amdgcn_wmma_f32_16x16x32_f16(false, a, false, b,
                                                (short)0, c, false, false);
}

// ---------------------------------------------------------------------------
// Generic WMMA GEMM: C[M,N] = A[M,K](f32) * B[N,K](f16)^T + bias + resid
// Register-blocked: one wave per 16x64 output tile (A fragment reused across
// 4 independent WMMAs per K-step). Requires N % 64 == 0 (true for all call
// sites: N in {128, 256}). K tail handled by zero-padded loads (K = 48).
// ---------------------------------------------------------------------------
template<bool OUT16>
__global__ __launch_bounds__(256)
void k_gemm(const float* __restrict__ A, const _Float16* __restrict__ B,
            const float* __restrict__ bias, const float* __restrict__ resid,
            void* __restrict__ Cout, int M, int N, int K, int relu)
{
  const int lane = threadIdx.x & 31;
  const int wave = threadIdx.x >> 5;
  const int half = lane >> 4;
  const int ln   = lane & 15;
  const int  ntn64  = N >> 6;                       // 16x64 tiles per row
  const long ntiles = (long)(M >> 4) * ntn64;
  long tile = (long)blockIdx.x * 8 + wave;
  if (tile >= ntiles) return;
  const long tm  = tile / ntn64;
  const int  tn0 = (int)(tile - tm * ntn64) * 4;    // first 16-col tile index

  const float* arow = A + (tm * 16 + ln) * (long)K;   // A: lane row = ln
  const _Float16* brow[4];
  #pragma unroll
  for (int j = 0; j < 4; ++j)
    brow[j] = B + (long)((tn0 + j) * 16 + ln) * K;    // B: lane col = ln

  v8f acc0 = {}, acc1 = {}, acc2 = {}, acc3 = {};
  for (int kk = 0; kk < K; kk += 32) {
    v16h a;
    if (kk + 32 <= K) {
      __builtin_prefetch(arow + kk + 64, 0, 1);
      #pragma unroll
      for (int i = 0; i < 8; ++i) {
        a[i]     = (_Float16)arow[kk + half * 16 + i];
        a[i + 8] = (_Float16)arow[kk + half * 16 + 8 + i];
      }
      v16h b0, b1, b2, b3;
      #pragma unroll
      for (int i = 0; i < 16; ++i) {
        b0[i] = brow[0][kk + half * 16 + i];
        b1[i] = brow[1][kk + half * 16 + i];
        b2[i] = brow[2][kk + half * 16 + i];
        b3[i] = brow[3][kk + half * 16 + i];
      }
      acc0 = wmma32(a, b0, acc0);
      acc1 = wmma32(a, b1, acc1);
      acc2 = wmma32(a, b2, acc2);
      acc3 = wmma32(a, b3, acc3);
    } else { // zero-padded K tail (e.g. K = 48)
      #pragma unroll
      for (int i = 0; i < 8; ++i) {
        int k1 = kk + half * 16 + i, k2 = kk + half * 16 + 8 + i;
        a[i]     = (k1 < K) ? (_Float16)arow[k1] : (_Float16)0.0f;
        a[i + 8] = (k2 < K) ? (_Float16)arow[k2] : (_Float16)0.0f;
      }
      v16h b0, b1, b2, b3;
      #pragma unroll
      for (int i = 0; i < 16; ++i) {
        int k = kk + half * 16 + i;
        _Float16 z = (_Float16)0.0f;
        b0[i] = (k < K) ? brow[0][k] : z;
        b1[i] = (k < K) ? brow[1][k] : z;
        b2[i] = (k < K) ? brow[2][k] : z;
        b3[i] = (k < K) ? brow[3][k] : z;
      }
      acc0 = wmma32(a, b0, acc0);
      acc1 = wmma32(a, b1, acc1);
      acc2 = wmma32(a, b2, acc2);
      acc3 = wmma32(a, b3, acc3);
    }
  }

  v8f accs[4] = {acc0, acc1, acc2, acc3};
  #pragma unroll
  for (int j = 0; j < 4; ++j) {
    const int col = (tn0 + j) * 16 + ln;
    const float bv = bias ? bias[col] : 0.0f;
    #pragma unroll
    for (int r = 0; r < 8; ++r) {
      long row = tm * 16 + r + 8 * half;   // C layout: VGPR r -> M = r + 8*half
      float v = accs[j][r] + bv;
      if (resid) v += resid[row * (long)N + col];
      if (relu)  v = fmaxf(v, 0.0f);
      if (OUT16) ((_Float16*)Cout)[row * (long)N + col] = (_Float16)v;
      else       ((float*)Cout)[row * (long)N + col] = v;
    }
  }
}

// ---------------------------------------------------------------------------
// Edge encoder: eh[e,0:128] = edge_attr[e,0:12] @ W[128,12]^T + b  (K=12 tiny)
// ---------------------------------------------------------------------------
__global__ __launch_bounds__(256)
void k_edge_enc(const float* __restrict__ ea, const float* __restrict__ W,
                const float* __restrict__ b, _Float16* __restrict__ eh, long total)
{
  long gid = (long)blockIdx.x * 256 + threadIdx.x;
  if (gid >= total) return;
  int  n = (int)(gid & 127);
  long e = gid >> 7;
  const float* a = ea + e * 12;
  const float* w = W + n * 12;
  float s = b[n];
  #pragma unroll
  for (int i = 0; i < 12; ++i) s += a[i] * w[i];
  eh[gid] = (_Float16)s;
}

// ---------------------------------------------------------------------------
// Fused GINE layer front-end, one workgroup per graph:
//   e = eh @ Wlin^T + blin ; m = relu(x[src] + e) ; agg[dst] += m (LDS atomics)
//   h = x + agg
// ---------------------------------------------------------------------------
__global__ __launch_bounds__(256)
void k_gine(const _Float16* __restrict__ eh, const _Float16* __restrict__ Wl,
            const float* __restrict__ bl, const float* __restrict__ x,
            const int* __restrict__ eidx, float* __restrict__ h,
            int in_d, int Etot)
{
  __shared__ float agg[64 * 128];
  const int g   = blockIdx.x;
  const int tid = threadIdx.x;
  const int cnt = 64 * in_d;
  for (int t = tid; t < cnt; t += 256) agg[t] = 0.0f;
  __syncthreads();

  const int lane = tid & 31, wave = tid >> 5, half = lane >> 4, ln = lane & 15;
  const int ntn = in_d >> 4;        // 3 (layer0) or 8
  const int e0  = g * 256;          // 256 contiguous edges per graph

  for (int t = wave; t < 16 * ntn; t += 8) {
    const int tm = t % 16, tn = t / 16;
    const _Float16* arow = eh + (long)(e0 + tm * 16 + ln) * 128;
    const _Float16* brow = Wl + (long)(tn * 16 + ln) * 128;
    v8f acc = {};
    #pragma unroll
    for (int kk = 0; kk < 128; kk += 32) {
      v16h a, b;
      #pragma unroll
      for (int i = 0; i < 8; ++i) {
        a[i]     = arow[kk + half * 16 + i];
        a[i + 8] = arow[kk + half * 16 + 8 + i];
      }
      #pragma unroll
      for (int i = 0; i < 16; ++i) b[i] = brow[kk + half * 16 + i];
      acc = wmma32(a, b, acc);
    }
    const int col = tn * 16 + ln;
    const float bv = bl[col];
    #pragma unroll
    for (int r = 0; r < 8; ++r) {
      int edge = e0 + tm * 16 + r + 8 * half;
      int src  = eidx[edge];
      int dst  = eidx[Etot + edge];
      float v = acc[r] + bv + x[(long)src * in_d + col];
      v = fmaxf(v, 0.0f);
      atomicAdd(&agg[(dst - g * 64) * in_d + col], v);  // ds_add_f32
    }
  }
  __syncthreads();
  for (int t = tid; t < cnt; t += 256) {
    int node = g * 64 + t / in_d;
    int c    = t % in_d;
    h[(long)node * in_d + c] = x[(long)node * in_d + c] + agg[t];
  }
}

// ---------------------------------------------------------------------------
// BatchNorm: stats (block partials + global f32 atomics), finalize, apply+relu
// ---------------------------------------------------------------------------
__global__ void k_zero(float* p, int n) {
  int i = blockIdx.x * blockDim.x + threadIdx.x;
  if (i < n) p[i] = 0.0f;
}

__global__ __launch_bounds__(256)
void k_bn_stats(const float* __restrict__ y, float* __restrict__ sum,
                float* __restrict__ sumsq, int M, int C, int rpb)
{
  int r0 = blockIdx.x * rpb;
  int r1 = r0 + rpb; if (r1 > M) r1 = M;
  for (int c = threadIdx.x; c < C; c += 256) {
    float s = 0.0f, q = 0.0f;
    for (int r = r0; r < r1; ++r) {
      float v = y[(long)r * C + c];
      s += v; q += v * v;
    }
    atomicAdd(&sum[c], s);
    atomicAdd(&sumsq[c], q);
  }
}

__global__ void k_bn_fin(const float* sum, const float* sumsq, const float* g,
                         const float* b, float* scale, float* shift, int M, int C)
{
  int c = blockIdx.x * blockDim.x + threadIdx.x;
  if (c >= C) return;
  float mu  = sum[c] / (float)M;
  float var = sumsq[c] / (float)M - mu * mu;
  float sc  = g[c] * rsqrtf(var + 1e-5f);
  scale[c] = sc;
  shift[c] = b[c] - mu * sc;
}

__global__ __launch_bounds__(256)
void k_bn_relu(const float* __restrict__ in, float* __restrict__ out,
               const float* __restrict__ scale, const float* __restrict__ shift,
               long n, int C)
{
  long i = (long)blockIdx.x * 256 + threadIdx.x;
  if (i >= n) return;
  int c = (int)(i % C);
  out[i] = fmaxf(in[i] * scale[c] + shift[c], 0.0f);
}

// ---------------------------------------------------------------------------
// Fused cross-attention core, one workgroup per (graph, direction).
// Q/K/V staged f16 in LDS; WMMA for QK^T (K=32) and P*V (K=64).
// LDS: 3*16KB + 8KB = 56KB.
// ---------------------------------------------------------------------------
__global__ __launch_bounds__(256)
void k_attn(const _Float16* __restrict__ Qd, const _Float16* __restrict__ Kd,
            const _Float16* __restrict__ Vd, const _Float16* __restrict__ Qa,
            const _Float16* __restrict__ Ka, const _Float16* __restrict__ Va,
            float* __restrict__ attD, float* __restrict__ attA)
{
  __shared__ _Float16 sQ[64 * 128];
  __shared__ _Float16 sK[64 * 128];
  __shared__ _Float16 sV[64 * 128];
  __shared__ _Float16 sP[64 * 64];

  const int g   = blockIdx.x >> 1;
  const int dir = blockIdx.x & 1;
  const _Float16 *Q, *Kp, *V; float* out;
  if (dir == 0) { Q = Qd; Kp = Ka; V = Va; out = attD; }   // don <- acc
  else          { Q = Qa; Kp = Kd; V = Vd; out = attA; }   // acc <- don

  const long base = (long)g * 64 * 128;
  for (int t = threadIdx.x; t < 64 * 128; t += 256) {
    sQ[t] = Q[base + t]; sK[t] = Kp[base + t]; sV[t] = V[base + t];
  }
  __syncthreads();

  const int lane = threadIdx.x & 31, wave = threadIdx.x >> 5;
  const int half = lane >> 4, ln = lane & 15;
  const float scl = 0.17677669529663687f;   // 1/sqrt(32)

  for (int hh = 0; hh < 4; ++hh) {
    // S = (Q_h K_h^T) * scl  -> sP (f16); 16 tiles over 8 waves
    for (int t = wave; t < 16; t += 8) {
      const int tm = t >> 2, tn = t & 3;
      v16h a, b;
      const _Float16* arow = sQ + (tm * 16 + ln) * 128 + hh * 32;
      #pragma unroll
      for (int i = 0; i < 8; ++i) {
        a[i]     = arow[half * 16 + i];
        a[i + 8] = arow[half * 16 + 8 + i];
      }
      const _Float16* brow = sK + (tn * 16 + ln) * 128 + hh * 32;
      #pragma unroll
      for (int i = 0; i < 16; ++i) b[i] = brow[half * 16 + i];
      v8f acc = {};
      acc = wmma32(a, b, acc);
      #pragma unroll
      for (int r = 0; r < 8; ++r) {
        int row = tm * 16 + r + 8 * half, col = tn * 16 + ln;
        sP[row * 64 + col] = (_Float16)(acc[r] * scl);
      }
    }
    __syncthreads();

    // softmax over rows (64 threads, 3 passes through LDS)
    if (threadIdx.x < 64) {
      const int row = threadIdx.x;
      float m = -1e30f;
      for (int j = 0; j < 64; ++j) m = fmaxf(m, (float)sP[row * 64 + j]);
      float s = 0.0f;
      for (int j = 0; j < 64; ++j) s += __expf((float)sP[row * 64 + j] - m);
      float inv = 1.0f / s;
      for (int j = 0; j < 64; ++j)
        sP[row * 64 + j] = (_Float16)(__expf((float)sP[row * 64 + j] - m) * inv);
    }
    __syncthreads();

    // att_h = P(64x64) @ V_h(64x32); 8 tiles, one per wave
    {
      const int tm = wave >> 1, tn = wave & 1;
      v8f acc = {};
      #pragma unroll
      for (int kk = 0; kk < 64; kk += 32) {
        v16h a, b;
        const _Float16* arow = sP + (tm * 16 + ln) * 64 + kk;
        #pragma unroll
        for (int i = 0; i < 8; ++i) {
          a[i]     = arow[half * 16 + i];
          a[i + 8] = arow[half * 16 + 8 + i];
        }
        const int col = tn * 16 + ln;
        #pragma unroll
        for (int i = 0; i < 16; ++i)
          b[i] = sV[(kk + half * 16 + i) * 128 + hh * 32 + col];
        acc = wmma32(a, b, acc);
      }
      #pragma unroll
      for (int r = 0; r < 8; ++r) {
        int row = tm * 16 + r + 8 * half, col = tn * 16 + ln;
        out[((long)g * 64 + row) * 128 + hh * 32 + col] = acc[r];
      }
    }
    __syncthreads();
  }
}

// ---------------------------------------------------------------------------
// LayerNorm over last dim (C=128): one wave32 per row, shfl_xor reduction.
// ---------------------------------------------------------------------------
__global__ __launch_bounds__(256)
void k_ln(const float* __restrict__ y, const float* __restrict__ g,
          const float* __restrict__ b, float* __restrict__ out, int C)
{
  const int wave = threadIdx.x >> 5, lane = threadIdx.x & 31;
  const long row = (long)blockIdx.x * 8 + wave;
  const float* yr = y + row * C;
  float s = 0.0f, q = 0.0f;
  for (int c = lane; c < C; c += 32) { float v = yr[c]; s += v; q += v * v; }
  #pragma unroll
  for (int o = 16; o > 0; o >>= 1) {
    s += __shfl_xor(s, o, 32);
    q += __shfl_xor(q, o, 32);
  }
  float mu  = s / (float)C;
  float var = q / (float)C - mu * mu;
  float inv = rsqrtf(var + 1e-5f);
  for (int c = lane; c < C; c += 32)
    out[row * C + c] = (yr[c] - mu) * inv * g[c] + b[c];
}

// ---------------------------------------------------------------------------
// Pooling: combined[g] = [don.mean, don.sum, acc.mean, acc.sum] (each 128)
// ---------------------------------------------------------------------------
__global__ __launch_bounds__(256)
void k_pool(const float* __restrict__ don, const float* __restrict__ acc,
            float* __restrict__ comb)
{
  const int g = blockIdx.x, t = threadIdx.x;
  const int c = t & 127;
  const float* src = (t < 128) ? don : acc;
  const long base = (long)g * 64 * 128;
  float s = 0.0f;
  for (int n = 0; n < 64; ++n) s += src[base + n * 128 + c];
  const int off = (t < 128) ? 0 : 256;
  comb[(long)g * 512 + off + c]       = s * (1.0f / 64.0f);
  comb[(long)g * 512 + off + 128 + c] = s;
}

// ---------------------------------------------------------------------------
// Head: [1024,128] @ [6,128]^T + b  (N=6, plain dot)
// ---------------------------------------------------------------------------
__global__ __launch_bounds__(256)
void k_head(const float* __restrict__ z, const float* __restrict__ W,
            const float* __restrict__ b, float* __restrict__ out)
{
  int gid = blockIdx.x * 256 + threadIdx.x;
  if (gid >= 1024 * 6) return;
  int o = gid % 6, m = gid / 6;
  const float* zr = z + (long)m * 128;
  const float* w  = W + (long)o * 128;
  float s = b[o];
  #pragma unroll 8
  for (int k = 0; k < 128; ++k) s += zr[k] * w[k];
  out[gid] = s;
}

__global__ __launch_bounds__(256)
void k_f2h(const float* __restrict__ in, _Float16* __restrict__ out, long n)
{
  long i = (long)blockIdx.x * 256 + threadIdx.x;
  if (i < n) out[i] = (_Float16)in[i];
}

// ===========================================================================
// Host orchestration
// ===========================================================================
static inline unsigned gemm_grid(long M, long N) {
  long t = (M / 16) * (N / 64);   // 16x64 tiles, 8 waves per block
  return (unsigned)((t + 7) / 8);
}

extern "C" void kernel_launch(void* const* d_in, const int* in_sizes, int n_in,
                              void* d_out, int out_size, void* d_ws, size_t ws_size,
                              hipStream_t stream)
{
  (void)in_sizes; (void)n_in; (void)out_size; (void)ws_size;
  const int Bg = 1024, Etot = 262144;
  const long Nd = (long)Bg * 64;          // 65536 nodes per set

  // ---- input pointers (depth-first insertion-order pytree flatten) --------
  const float* x_don = (const float*)d_in[0];
  const float* x_acc = (const float*)d_in[1];
  const float* ea_d  = (const float*)d_in[2];
  const float* ea_a  = (const float*)d_in[3];
  const int*   ei_d  = (const int*)d_in[4];
  const int*   ei_a  = (const int*)d_in[5];
  const float* eenc_w = (const float*)d_in[6];
  const float* eenc_b = (const float*)d_in[7];
  auto LP = [&](int l, int j) { return (const float*)d_in[8 + l * 10 + j]; };
  // per-layer: 0 lin.w, 1 lin.b, 2 mlp1.w, 3 mlp1.b, 4 bn1.g, 5 bn1.b,
  //            6 mlp2.w, 7 mlp2.b, 8 bn2.g, 9 bn2.b
  const float* qw = (const float*)d_in[38]; const float* qb = (const float*)d_in[39];
  const float* kw = (const float*)d_in[40]; const float* kb = (const float*)d_in[41];
  const float* vw = (const float*)d_in[42]; const float* vb = (const float*)d_in[43];
  const float* ow = (const float*)d_in[44]; const float* ob = (const float*)d_in[45];
  const float* lng = (const float*)d_in[46]; const float* lnb = (const float*)d_in[47];
  const float* p1w = (const float*)d_in[48]; const float* p1b = (const float*)d_in[49];
  const float* p2w = (const float*)d_in[50]; const float* p2b = (const float*)d_in[51];
  const float* hw  = (const float*)d_in[52]; const float* hb  = (const float*)d_in[53];

  // ---- workspace layout ----------------------------------------------------
  char* ws = (char*)d_ws;
  size_t off = 0;
  auto alloc = [&](size_t bytes) -> size_t {
    size_t o = (off + 255) & ~(size_t)255;
    off = o + bytes;
    return o;
  };
  const int ind[3] = {48, 128, 128};
  size_t o_lin[3], o_m1[3], o_m2[3];
  for (int l = 0; l < 3; ++l) {
    o_lin[l] = alloc((size_t)ind[l] * 128 * 2);
    o_m1[l]  = alloc((size_t)256 * ind[l] * 2);
    o_m2[l]  = alloc((size_t)128 * 256 * 2);
  }
  size_t o_qw = alloc(128 * 128 * 2), o_kw = alloc(128 * 128 * 2);
  size_t o_vw = alloc(128 * 128 * 2), o_ow = alloc(128 * 128 * 2);
  size_t o_p1 = alloc(256 * 512 * 2), o_p2 = alloc(128 * 256 * 2);
  size_t o_ehD = alloc((size_t)Etot * 128 * 2);
  size_t o_ehA = alloc((size_t)Etot * 128 * 2);
  size_t o_xD  = alloc((size_t)Nd * 128 * 4);
  size_t o_xA  = alloc((size_t)Nd * 128 * 4);
  size_t o_T1  = alloc((size_t)Nd * 256 * 4);   // y1 / [Yd|Ya]
  size_t o_T2  = alloc((size_t)Nd * 128 * 4);   // h / y2 / comb,z1,z2
  size_t o_Q[2], o_K[2], o_V[2];
  for (int s = 0; s < 2; ++s) {
    o_Q[s] = alloc((size_t)Nd * 128 * 2);
    o_K[s] = alloc((size_t)Nd * 128 * 2);
    o_V[s] = alloc((size_t)Nd * 128 * 2);
  }
  size_t o_attD = alloc((size_t)Nd * 128 * 4);
  size_t o_attA = alloc((size_t)Nd * 128 * 4);
  size_t o_stat = alloc(4 * 256 * 4);

  float* sum   = (float*)(ws + o_stat);
  float* sumsq = sum + 256;
  float* scale = sum + 512;
  float* shift = sum + 768;

  auto f2h = [&](const float* src, size_t dst, long n) {
    k_f2h<<<(unsigned)((n + 255) / 256), 256, 0, stream>>>(src, (_Float16*)(ws + dst), n);
  };

  // ---- weight conversion to f16 -------------------------------------------
  for (int l = 0; l < 3; ++l) {
    f2h(LP(l, 0), o_lin[l], (long)ind[l] * 128);
    f2h(LP(l, 2), o_m1[l],  (long)256 * ind[l]);
    f2h(LP(l, 6), o_m2[l],  128 * 256);
  }
  f2h(qw, o_qw, 128 * 128); f2h(kw, o_kw, 128 * 128);
  f2h(vw, o_vw, 128 * 128); f2h(ow, o_ow, 128 * 128);
  f2h(p1w, o_p1, 256 * 512); f2h(p2w, o_p2, 128 * 256);

  // ---- edge encoder (shared across layers) --------------------------------
  {
    long tot = (long)Etot * 128;
    unsigned g = (unsigned)((tot + 255) / 256);
    k_edge_enc<<<g, 256, 0, stream>>>(ea_d, eenc_w, eenc_b, (_Float16*)(ws + o_ehD), tot);
    k_edge_enc<<<g, 256, 0, stream>>>(ea_a, eenc_w, eenc_b, (_Float16*)(ws + o_ehA), tot);
  }

  auto bn = [&](const float* y, int M, int C, const float* g, const float* b) {
    k_zero<<<2, 256, 0, stream>>>(sum, 512);
    k_bn_stats<<<M / 256, 256, 0, stream>>>(y, sum, sumsq, M, C, 256);
    k_bn_fin<<<1, 256, 0, stream>>>(sum, sumsq, g, b, scale, shift, M, C);
  };

  float* T1 = (float*)(ws + o_T1);
  float* T2 = (float*)(ws + o_T2);

  // ---- 3 GINE layers, both node sets (shared params, shared temps) --------
  for (int l = 0; l < 3; ++l) {
    const int in_d = ind[l];
    for (int s = 0; s < 2; ++s) {
      const float* xin = (l == 0) ? (s == 0 ? x_don : x_acc)
                                  : (const float*)(ws + (s == 0 ? o_xD : o_xA));
      const _Float16* eh = (const _Float16*)(ws + (s == 0 ? o_ehD : o_ehA));
      const int* eidx = (s == 0) ? ei_d : ei_a;
      float* Xout = (float*)(ws + (s == 0 ? o_xD : o_xA));

      // h = x + segment_sum(relu(x[src] + eh@Wlin^T + b), dst)
      k_gine<<<Bg, 256, 0, stream>>>(eh, (const _Float16*)(ws + o_lin[l]),
                                     LP(l, 1), xin, eidx, T2, in_d, Etot);
      // y1 = h @ mlp1^T + b1
      k_gemm<false><<<gemm_grid(Nd, 256), 256, 0, stream>>>(
          T2, (const _Float16*)(ws + o_m1[l]), LP(l, 3), nullptr, T1,
          (int)Nd, 256, in_d, 0);
      bn(T1, (int)Nd, 256, LP(l, 4), LP(l, 5));
      k_bn_relu<<<(unsigned)((Nd * 256 + 255) / 256), 256, 0, stream>>>(
          T1, T1, scale, shift, Nd * 256, 256);
      // y2 = z @ mlp2^T + b2
      k_gemm<false><<<gemm_grid(Nd, 128), 256, 0, stream>>>(
          T1, (const _Float16*)(ws + o_m2[l]), LP(l, 7), nullptr, T2,
          (int)Nd, 128, 256, 0);
      bn(T2, (int)Nd, 128, LP(l, 8), LP(l, 9));
      k_bn_relu<<<(unsigned)((Nd * 128 + 255) / 256), 256, 0, stream>>>(
          T2, Xout, scale, shift, Nd * 128, 128);
    }
  }

  // ---- cross attention ----------------------------------------------------
  for (int s = 0; s < 2; ++s) {
    const float* X = (const float*)(ws + (s == 0 ? o_xD : o_xA));
    k_gemm<true><<<gemm_grid(Nd, 128), 256, 0, stream>>>(
        X, (const _Float16*)(ws + o_qw), qb, nullptr, ws + o_Q[s], (int)Nd, 128, 128, 0);
    k_gemm<true><<<gemm_grid(Nd, 128), 256, 0, stream>>>(
        X, (const _Float16*)(ws + o_kw), kb, nullptr, ws + o_K[s], (int)Nd, 128, 128, 0);
    k_gemm<true><<<gemm_grid(Nd, 128), 256, 0, stream>>>(
        X, (const _Float16*)(ws + o_vw), vb, nullptr, ws + o_V[s], (int)Nd, 128, 128, 0);
  }
  k_attn<<<2 * Bg, 256, 0, stream>>>(
      (const _Float16*)(ws + o_Q[0]), (const _Float16*)(ws + o_K[0]),
      (const _Float16*)(ws + o_V[0]), (const _Float16*)(ws + o_Q[1]),
      (const _Float16*)(ws + o_K[1]), (const _Float16*)(ws + o_V[1]),
      (float*)(ws + o_attD), (float*)(ws + o_attA));

  // o-projection + residual, then LayerNorm (final features overwrite X)
  float* Yd = T1;
  float* Ya = T1 + Nd * 128;
  k_gemm<false><<<gemm_grid(Nd, 128), 256, 0, stream>>>(
      (const float*)(ws + o_attD), (const _Float16*)(ws + o_ow), ob,
      (const float*)(ws + o_xD), Yd, (int)Nd, 128, 128, 0);
  k_gemm<false><<<gemm_grid(Nd, 128), 256, 0, stream>>>(
      (const float*)(ws + o_attA), (const _Float16*)(ws + o_ow), ob,
      (const float*)(ws + o_xA), Ya, (int)Nd, 128, 128, 0);
  k_ln<<<(unsigned)(Nd / 8), 256, 0, stream>>>(Yd, lng, lnb, (float*)(ws + o_xD), 128);
  k_ln<<<(unsigned)(Nd / 8), 256, 0, stream>>>(Ya, lng, lnb, (float*)(ws + o_xA), 128);

  // ---- pooling + MLP head -------------------------------------------------
  float* comb = T2;                         // [1024, 512]
  float* z1   = T2 + (long)Bg * 512;        // [1024, 256]
  float* z2   = z1 + (long)Bg * 256;        // [1024, 128]
  k_pool<<<Bg, 256, 0, stream>>>((const float*)(ws + o_xD), (const float*)(ws + o_xA), comb);
  k_gemm<false><<<gemm_grid(Bg, 256), 256, 0, stream>>>(
      comb, (const _Float16*)(ws + o_p1), p1b, nullptr, z1, Bg, 256, 512, 1);
  k_gemm<false><<<gemm_grid(Bg, 128), 256, 0, stream>>>(
      z1, (const _Float16*)(ws + o_p2), p2b, nullptr, z2, Bg, 128, 256, 1);
  k_head<<<(1024 * 6 + 255) / 256, 256, 0, stream>>>(z2, hw, hb, (float*)d_out);
}
